// DKVMN_6365141533206
// MI455X (gfx1250) — matrix operand order repaired and checked by reading, not compiled, not run on hardware
//
#include <hip/hip_runtime.h>

// ---------------------------------------------------------------------------
// DKVMN forward for MI455X (gfx1250, wave32).
//
// * Batch-parallel GEMMs (attention logits+softmax, erase/add linears) hoisted
//   out of the T-loop, run as v_wmma_f32_16x16x32_f16 over M = B*T = 131072.
// * Sequential recurrence keeps each batch element's 512x256 value memory
//   resident in LDS as fp16 (256 KB of the 320 KB WGP LDS), one workgroup per
//   batch element -> removes ~137 GB of HBM state traffic (5.9 ms @ 23.3TB/s).
// * Per-step w/erase/add streams (2 KB/step) are double-buffered into LDS via
//   CDNA5 async copies (global_load_async_to_lds_b128 + s_wait_asynccnt) so
//   global latency overlaps the LDS compute pass.
// * Hot loop: thread owns 4 fp16 columns (b64 LDS accesses) and 1/4 of the
//   n-range; w fetched 8-at-a-time with b128 -> ~272 DS ops/thread/step.
// ---------------------------------------------------------------------------

typedef __attribute__((ext_vector_type(16))) _Float16 v16h;
typedef __attribute__((ext_vector_type(8)))  _Float16 v8h;
typedef __attribute__((ext_vector_type(4)))  _Float16 h4;
typedef __attribute__((ext_vector_type(8)))  float    v8f;

#define B_DIM 256
#define T_DIM 512
#define N_DIM 512
#define DK    64
#define DV    256
#define BT    (B_DIM * T_DIM)   // 131072

// ===========================================================================
// Kernel 1: logits = Q @ key_memory^T, fused row softmax -> w (fp16).
// One block = 16 rows x 512 cols; 8 waves x 4 n-tiles x 2 k-chunks WMMAs.
// ===========================================================================
__global__ __launch_bounds__(256) void logits_softmax_kernel(
    const int*   __restrict__ concepts,       // (BT)
    const float* __restrict__ concept_embed,  // (N+1, DK), row 0 == 0
    const float* __restrict__ key_memory,     // (N, DK)
    _Float16*    __restrict__ g_w)            // (BT, N)
{
  __shared__ float qs[16][DK];
  __shared__ float logits[16][N_DIM];
  __shared__ float red[16][16];
  __shared__ float rowmax[16], rowsum[16];

  const int m0  = blockIdx.x * 16;
  const int tid = threadIdx.x;

  for (int idx = tid; idx < 16 * DK; idx += 256) {
    int row = idx >> 6, k = idx & 63;
    int c = concepts[m0 + row];
    qs[row][k] = (c != 0) ? concept_embed[c * DK + k] : 0.f;
  }
  __syncthreads();

  const int lane = tid & 31;
  const int wave = tid >> 5;
  const int half = lane >> 4;   // K-half selector per ISA A/B layouts
  const int l16  = lane & 15;

  for (int tile = 0; tile < 4; ++tile) {
    const int n0 = wave * 64 + tile * 16;
    v8f acc = {};
    for (int kc = 0; kc < 2; ++kc) {
      const int k0 = kc * 32;
      v16h a, bf;
      #pragma unroll
      for (int j = 0; j < 16; ++j) {
        int Ka = k0 + half * 16 + j;   // A 16x32 layout
        a[j] = (_Float16)qs[l16][Ka];
        int Kb = k0 + half * 16 + j;                        // B 32x16 layout
        bf[j] = (_Float16)key_memory[(n0 + l16) * DK + Kb];
      }
      acc = __builtin_amdgcn_wmma_f32_16x16x32_f16(
          false, a, false, bf, (short)0, acc, false, false);
    }
    #pragma unroll
    for (int r = 0; r < 8; ++r)
      logits[r + 8 * half][n0 + l16] = acc[r];   // D: M = r + 8*half, N = l16
  }
  __syncthreads();

  const int row = tid >> 4, seg = tid & 15;
  float lm = -3.4e38f;
  for (int j = 0; j < 32; ++j) lm = fmaxf(lm, logits[row][seg * 32 + j]);
  red[row][seg] = lm;
  __syncthreads();
  if (seg == 0) {
    float m = red[row][0];
    for (int j = 1; j < 16; ++j) m = fmaxf(m, red[row][j]);
    rowmax[row] = m;
  }
  __syncthreads();
  const float rm = rowmax[row];
  float ls = 0.f;
  for (int j = 0; j < 32; ++j) ls += __expf(logits[row][seg * 32 + j] - rm);
  red[row][seg] = ls;
  __syncthreads();
  if (seg == 0) {
    float s = 0.f;
    for (int j = 0; j < 16; ++j) s += red[row][j];
    rowsum[row] = s;
  }
  __syncthreads();
  const float inv = 1.f / rowsum[row];
  for (int j = 0; j < 32; ++j) {
    int col = seg * 32 + j;
    g_w[(size_t)(m0 + row) * N_DIM + col] =
        (_Float16)(__expf(logits[row][col] - rm) * inv);
  }
}

// ===========================================================================
// Kernel 2: erase = sigmoid(V@We^T + be), add = tanh(V@Wa^T + ba).
// One block = 16 rows x 256 cols for BOTH outputs; 32 WMMAs per wave.
// ===========================================================================
__global__ __launch_bounds__(256) void erase_add_kernel(
    const int*   __restrict__ interactions,      // (BT)
    const float* __restrict__ interaction_embed, // (2N+1, DV), row 0 == 0
    const float* __restrict__ We, const float* __restrict__ be,
    const float* __restrict__ Wa, const float* __restrict__ ba,
    _Float16* __restrict__ g_e,                  // (BT, DV)
    _Float16* __restrict__ g_a)                  // (BT, DV)
{
  __shared__ float vs[16][DV];
  const int m0  = blockIdx.x * 16;
  const int tid = threadIdx.x;

  for (int idx = tid; idx < 16 * DV; idx += 256) {
    int row = idx >> 8, k = idx & 255;
    int iid = interactions[m0 + row];
    vs[row][k] = (iid != 0) ? interaction_embed[iid * DV + k] : 0.f;
  }
  __syncthreads();

  const int lane = tid & 31;
  const int wave = tid >> 5;
  const int half = lane >> 4;
  const int l16  = lane & 15;

  v8f accE[2] = {}, accA[2] = {};
  for (int kc = 0; kc < 8; ++kc) {
    const int k0 = kc * 32;
    v16h a;
    #pragma unroll
    for (int j = 0; j < 16; ++j) {
      int Ka = k0 + half * 16 + j;
      a[j] = (_Float16)vs[l16][Ka];
    }
    #pragma unroll
    for (int tile = 0; tile < 2; ++tile) {
      const int n0 = wave * 32 + tile * 16;
      v16h bE, bA;
      #pragma unroll
      for (int j = 0; j < 16; ++j) {
        int Kb = k0 + half * 16 + j;   // contiguous -> b128 loads
        bE[j] = (_Float16)We[(n0 + l16) * DV + Kb];
        bA[j] = (_Float16)Wa[(n0 + l16) * DV + Kb];
      }
      accE[tile] = __builtin_amdgcn_wmma_f32_16x16x32_f16(
          false, a, false, bE, (short)0, accE[tile], false, false);
      accA[tile] = __builtin_amdgcn_wmma_f32_16x16x32_f16(
          false, a, false, bA, (short)0, accA[tile], false, false);
    }
  }
  #pragma unroll
  for (int tile = 0; tile < 2; ++tile) {
    const int n = wave * 32 + tile * 16 + l16;
    #pragma unroll
    for (int r = 0; r < 8; ++r) {
      const int m = m0 + r + 8 * half;
      float e  = accE[tile][r] + be[n];
      float aa = accA[tile][r] + ba[n];
      e  = 1.f / (1.f + __expf(-e));
      aa = tanhf(aa);
      g_e[(size_t)m * DV + n] = (_Float16)e;
      g_a[(size_t)m * DV + n] = (_Float16)aa;
    }
  }
}

// ===========================================================================
// CDNA5 async global->LDS copy (ASYNCcnt-tracked). Inline asm per bridge doc
// (builtin arity differs across toolchains). GV mode: 64-bit vaddr, no saddr.
// ===========================================================================
static __device__ __forceinline__ void async_g2l_b128(unsigned lds_off,
                                                      const void* gptr) {
  asm volatile("global_load_async_to_lds_b128 %0, %1, off"
               :: "v"(lds_off), "v"((unsigned long long)(uintptr_t)gptr)
               : "memory");
}
static __device__ __forceinline__ void wait_async0() {
  asm volatile("s_wait_asynccnt 0x0" ::: "memory");
}

// LDS layout (bytes) for the recurrence kernel:
#define OFF_MEM   0                       // _Float16 [512][256]   262144 B
#define OFF_W     (OFF_MEM + N_DIM * DV * 2)      // _Float16 [2][512]   2048 B
#define OFF_E     (OFF_W + 2 * N_DIM * 2)         // _Float16 [2][256]   1024 B
#define OFF_A     (OFF_E + 2 * DV * 2)            // _Float16 [2][256]   1024 B
#define OFF_RP    (OFF_A + 2 * DV * 2)            // float [4][256]      4096 B
#define OFF_Q     (OFF_RP + 4 * DV * 4)           // float [64]           256 B
#define OFF_H1    (OFF_Q + DK * 4)                // float [64]           256 B
#define OFF_H2    (OFF_H1 + 64 * 4)               // float [64]           256 B
#define SMEM_BYTES (OFF_H2 + 64 * 4)              // 271104 B

// ===========================================================================
// Kernel 3: sequential recurrence. One block per batch element; value memory
// fp16-resident in LDS. Thread owns 4 adjacent columns (b64 LDS access) and
// 1/4 of the n-range; per-step w/e/a double-buffered via async copies.
// ===========================================================================
__global__ __launch_bounds__(256) void dkvmn_recurrence_kernel(
    const int*      __restrict__ concepts,
    const float*    __restrict__ value_memory,   // (N, DV)
    const float*    __restrict__ concept_embed,  // (N+1, DK)
    const _Float16* __restrict__ g_w,            // (BT, N)
    const _Float16* __restrict__ g_e,            // (BT, DV)
    const _Float16* __restrict__ g_a,            // (BT, DV)
    const float* __restrict__ W1, const float* __restrict__ b1,  // (64,320)
    const float* __restrict__ W2, const float* __restrict__ b2,  // (64,64)
    const float* __restrict__ W3, const float* __restrict__ b3,  // (64),(1)
    float* __restrict__ out)                     // (B, T)
{
  extern __shared__ char smem[];
  _Float16* mem   = (_Float16*)(smem + OFF_MEM);   // [n*256 + d]
  _Float16* w_lds = (_Float16*)(smem + OFF_W);     // [buf*512 + n]
  _Float16* e_lds = (_Float16*)(smem + OFF_E);     // [buf*256 + d]
  _Float16* a_lds = (_Float16*)(smem + OFF_A);     // [buf*256 + d]
  float*    rp    = (float*)   (smem + OFF_RP);    // [nq*256 + d]
  float*    q_s   = (float*)   (smem + OFF_Q);
  float*    h1_s  = (float*)   (smem + OFF_H1);
  float*    h2_s  = (float*)   (smem + OFF_H2);
  const unsigned lds_base = (unsigned)(uintptr_t)smem;

  const int b   = blockIdx.x;
  const int tid = threadIdx.x;
  const int c4  = (tid & 63) * 4;   // first of 4 owned columns
  const int nq  = tid >> 6;         // n-quarter: [nq*128, nq*128+128)

  // Issue async prefetch of step-0 w/e/a into buffer 0 (overlaps mem init).
  // Roles: tid 0..63 -> w (64x16B), 64..95 -> e (32x16B), 96..127 -> a.
  {
    const size_t s0 = (size_t)b * T_DIM;
    if (tid < 64) {
      async_g2l_b128(lds_base + OFF_W + tid * 16,
                     (const char*)(g_w + s0 * N_DIM) + tid * 16);
    } else if (tid < 96) {
      async_g2l_b128(lds_base + OFF_E + (tid - 64) * 16,
                     (const char*)(g_e + s0 * DV) + (tid - 64) * 16);
    } else if (tid < 128) {
      async_g2l_b128(lds_base + OFF_A + (tid - 96) * 16,
                     (const char*)(g_a + s0 * DV) + (tid - 96) * 16);
    }
  }

  // Broadcast-init memory state into LDS as fp16.
  for (int idx = tid; idx < N_DIM * DV; idx += 256)
    mem[idx] = (_Float16)value_memory[idx];
  wait_async0();
  __syncthreads();

  for (int t = 0; t < T_DIM; ++t) {
    const size_t step = (size_t)b * T_DIM + t;
    const int cur = t & 1, nxt = cur ^ 1;
    const _Float16* w_cur = w_lds + cur * N_DIM;

    // Per-thread erase/add for the 4 owned columns (valid from prior wait).
    h4 ev4 = *(const h4*)(e_lds + cur * DV + c4);
    h4 av4 = *(const h4*)(a_lds + cur * DV + c4);
    const float e0 = (float)ev4[0], e1 = (float)ev4[1],
                e2 = (float)ev4[2], e3 = (float)ev4[3];
    const float a0 = (float)av4[0], a1 = (float)av4[1],
                a2 = (float)av4[2], a3 = (float)av4[3];
    if (tid < DK) {
      int c = concepts[step];
      q_s[tid] = (c != 0) ? concept_embed[c * DK + tid] : 0.f;
    }

    // Kick off async prefetch of step t+1 into the other buffers.
    if (t + 1 < T_DIM) {
      const size_t s1 = step + 1;
      if (tid < 64) {
        async_g2l_b128(lds_base + OFF_W + nxt * N_DIM * 2 + tid * 16,
                       (const char*)(g_w + s1 * N_DIM) + tid * 16);
      } else if (tid < 96) {
        async_g2l_b128(lds_base + OFF_E + nxt * DV * 2 + (tid - 64) * 16,
                       (const char*)(g_e + s1 * DV) + (tid - 64) * 16);
      } else if (tid < 128) {
        async_g2l_b128(lds_base + OFF_A + nxt * DV * 2 + (tid - 96) * 16,
                       (const char*)(g_a + s1 * DV) + (tid - 96) * 16);
      }
    }

    // Fused read (pre-update) + erase/add update.
    // Columns [c4, c4+3] x n in [nq*128, nq*128+128). b64 LDS accesses hit
    // all 64 banks exactly once per wave; w fetched 8-at-a-time via b128.
    float r0 = 0.f, r1 = 0.f, r2 = 0.f, r3 = 0.f;
    const int nbase = nq * 128;
    for (int g = 0; g < 16; ++g) {
      v8h wv = *(const v8h*)(w_cur + nbase + g * 8);
      #pragma unroll
      for (int u = 0; u < 8; ++u) {
        const int n = nbase + g * 8 + u;
        const float wn = (float)wv[u];
        h4 m = *(const h4*)(mem + n * DV + c4);
        const float m0 = (float)m[0], m1 = (float)m[1],
                    m2 = (float)m[2], m3 = (float)m[3];
        r0 += wn * m0; r1 += wn * m1; r2 += wn * m2; r3 += wn * m3;
        h4 o;
        o[0] = (_Float16)(m0 * (1.f - wn * e0) + wn * a0);
        o[1] = (_Float16)(m1 * (1.f - wn * e1) + wn * a1);
        o[2] = (_Float16)(m2 * (1.f - wn * e2) + wn * a2);
        o[3] = (_Float16)(m3 * (1.f - wn * e3) + wn * a3);
        *(h4*)(mem + n * DV + c4) = o;
      }
    }
    {
      typedef float f4v __attribute__((ext_vector_type(4)));
      f4v rv; rv[0] = r0; rv[1] = r1; rv[2] = r2; rv[3] = r3;
      *(f4v*)(rp + nq * DV + c4) = rv;
    }
    __syncthreads();

    // Small MLP (fp32): [r,q](320) -> 64 -> 64 -> 1.
    if (tid < 64) {
      float acc = b1[tid];
      const float* w1r = W1 + tid * (DV + DK);
      for (int k = 0; k < DV; ++k) {
        float rk = rp[k] + rp[DV + k] + rp[2 * DV + k] + rp[3 * DV + k];
        acc += rk * w1r[k];
      }
      for (int k = 0; k < DK; ++k) acc += q_s[k] * w1r[DV + k];
      h1_s[tid] = fmaxf(acc, 0.f);
    }
    __syncthreads();
    if (tid < 64) {
      float acc = b2[tid];
      const float* w2r = W2 + tid * 64;
      for (int k = 0; k < 64; ++k) acc += h1_s[k] * w2r[k];
      h2_s[tid] = fmaxf(acc, 0.f);
    }
    __syncthreads();
    if (tid == 0) {
      float acc = b3[0];
      for (int k = 0; k < 64; ++k) acc += h2_s[k] * W3[k];
      out[step] = 1.f / (1.f + __expf(-acc));
    }
    wait_async0();      // step t+1 buffers resident
    __syncthreads();    // visible to all waves; protects q_s/rp/h*
  }
}

// ===========================================================================
// Launch
// ===========================================================================
extern "C" void kernel_launch(void* const* d_in, const int* in_sizes, int n_in,
                              void* d_out, int out_size, void* d_ws, size_t ws_size,
                              hipStream_t stream) {
  (void)in_sizes; (void)n_in; (void)out_size; (void)ws_size;

  const int*   concepts          = (const int*)  d_in[0];
  const int*   interactions      = (const int*)  d_in[1];
  const float* key_memory        = (const float*)d_in[2];
  const float* value_memory      = (const float*)d_in[3];
  const float* concept_embed     = (const float*)d_in[4];
  const float* interaction_embed = (const float*)d_in[5];
  const float* We = (const float*)d_in[6];
  const float* be = (const float*)d_in[7];
  const float* Wa = (const float*)d_in[8];
  const float* ba = (const float*)d_in[9];
  const float* W1 = (const float*)d_in[10];
  const float* b1 = (const float*)d_in[11];
  const float* W2 = (const float*)d_in[12];
  const float* b2 = (const float*)d_in[13];
  const float* W3 = (const float*)d_in[14];
  const float* b3 = (const float*)d_in[15];
  float* out = (float*)d_out;

  // Workspace: w (134 MB) + erase (67 MB) + add (67 MB), all fp16.
  char* ws = (char*)d_ws;
  _Float16* g_w = (_Float16*)ws;
  _Float16* g_e = (_Float16*)(ws + (size_t)BT * N_DIM * 2);
  _Float16* g_a = (_Float16*)(ws + (size_t)BT * N_DIM * 2 + (size_t)BT * DV * 2);

  logits_softmax_kernel<<<BT / 16, 256, 0, stream>>>(
      concepts, concept_embed, key_memory, g_w);
  erase_add_kernel<<<BT / 16, 256, 0, stream>>>(
      interactions, interaction_embed, We, be, Wa, ba, g_e, g_a);

  hipFuncSetAttribute((const void*)dkvmn_recurrence_kernel,
                      hipFuncAttributeMaxDynamicSharedMemorySize, SMEM_BYTES);
  dkvmn_recurrence_kernel<<<B_DIM, 256, SMEM_BYTES, stream>>>(
      concepts, value_memory, concept_embed, g_w, g_e, g_a,
      W1, b1, W2, b2, W3, b3, out);
}
